// SparseTopKMoE_4801773437213
// MI455X (gfx1250) — matrix operand order repaired and checked by reading, not compiled, not run on hardware
//
#include <hip/hip_runtime.h>
#include <cmath>

// ---- problem constants (fixed by the reference) ----
#define B_DIM   8
#define C_DIM   192
#define H_DIM   128
#define W_DIM   128
#define E_DIM   4
#define HID_DIM 384
#define HW      (H_DIM * W_DIM)

// ---- tiling ----
#define MT         128            // tokens per block
#define NTHREADS   256            // 8 waves (wave32)
#define TOK_STRIDE 200            // bf16 elems per token row (192 + pad), 400B (16B mult)
#define HID_STRIDE 392            // bf16 elems per hid row (384 + pad), 784B (16B mult)
#define OUT_STRIDE 196            // f32 elems per out row; same bytes as HID_STRIDE bf16

#define W_ELEMS (E_DIM * HID_DIM * C_DIM)   // 294912 elements (same for fc1 and fc2)

typedef __attribute__((ext_vector_type(16))) __bf16          v16bf;
typedef __attribute__((ext_vector_type(8)))  float           v8f;
typedef __attribute__((ext_vector_type(8)))  unsigned short  v8us;

__device__ __forceinline__ unsigned short f2bf(float f) {
  unsigned int u = __float_as_uint(f);
  u += 0x7FFFu + ((u >> 16) & 1u);      // round-to-nearest-even truncation
  return (unsigned short)(u >> 16);
}
__device__ __forceinline__ float bf2f(unsigned short s) {
  return __uint_as_float(((unsigned int)s) << 16);
}
__device__ __forceinline__ float gelu_exact(float x) {
  return 0.5f * x * (1.0f + erff(x * 0.70710678118654752f));
}

union Frag16 { v8us u[2]; v16bf b; };

// A-matrix 16x32 bf16 fragment from row-major LDS tile.
// ISA layout: lanes 0-15 -> M=lane, K={kb+0..7, kb+16..23};
//             lanes 16-31 -> M=lane-16, K={kb+8..15, kb+24..31}
__device__ __forceinline__ v16bf load_afrag_lds(const unsigned short* lds, int row0,
                                                int stride, int kb, int lane) {
  const unsigned short* p = lds + (row0 + (lane & 15)) * stride + kb + ((lane >> 4) << 3);
  Frag16 f;
  f.u[0] = *(const v8us*)(p);        // K = base .. base+7
  f.u[1] = *(const v8us*)(p + 16);   // K = base+16 .. base+23
  return f.b;
}

// B-matrix 32x16 bf16 fragment from pre-converted bf16 weights, (N, K) row-major.
// Lane layout: lanes 0-15 -> N=lane, K={kb+0..7, kb+16..23}; lanes 16-31 -> N=lane-16, K={kb+8..15, kb+24..31}.
// Two global_load_b128, zero conversion VALU.
__device__ __forceinline__ v16bf load_bfrag_bf(const unsigned short* __restrict__ w, int n0,
                                               int rowStride, int kb, int lane) {
  const unsigned short* p = w + (size_t)(n0 + (lane & 15)) * rowStride + kb + ((lane >> 4) << 3);
  Frag16 f;
  f.u[0] = *(const v8us*)(p);
  f.u[1] = *(const v8us*)(p + 16);
  return f.b;
}

#define WMMA_BF16(A, B, C) \
  __builtin_amdgcn_wmma_f32_16x16x32_bf16(false, (A), false, (B), (short)0, (C), false, false)

// ---- prep kernel: fp32 weights -> bf16 copies in workspace ----
__global__ void __launch_bounds__(256)
convert_weights_kernel(const float* __restrict__ fc1_w, const float* __restrict__ fc2_w,
                       unsigned short* __restrict__ w1bf, unsigned short* __restrict__ w2bf) {
  int i = blockIdx.x * 256 + threadIdx.x;
  if (i < W_ELEMS) {
    w1bf[i] = f2bf(fc1_w[i]);
    w2bf[i] = f2bf(fc2_w[i]);
  }
}

__global__ void __launch_bounds__(NTHREADS, 1)
moe_topk_kernel(const float* __restrict__ x,
                const float* __restrict__ router_w, const float* __restrict__ router_b,
                const unsigned short* __restrict__ w1bf, const float* __restrict__ fc1_b,
                const unsigned short* __restrict__ w2bf, const float* __restrict__ fc2_b,
                const float* __restrict__ scale_p,
                float* __restrict__ out)
{
  extern __shared__ __align__(16) unsigned short smem[];
  unsigned short* tokLds = smem;                                     // MT*TOK_STRIDE bf16
  unsigned short* hidLds = smem + MT * TOK_STRIDE;                   // MT*HID_STRIDE bf16
  float* gatesLds = (float*)(smem + MT * (TOK_STRIDE + HID_STRIDE)); // MT*4 f32
  float* outLds   = (float*)hidLds;                                  // reuse: MT*OUT_STRIDE f32

  const int tid  = threadIdx.x;
  const int lane = tid & 31;
  const int wv   = tid >> 5;                  // wave id 0..7 == M-tile id

  const int gt0  = blockIdx.x * MT;           // first token of this tile
  const int bimg = gt0 / HW;                  // tile never crosses an image (HW % MT == 0)
  const int p0   = gt0 - bimg * HW;
  const float* xb = x   + (size_t)bimg * C_DIM * HW + p0;
  float*       ob = out + (size_t)bimg * C_DIM * HW + p0;

  // ---- 1) stage token tile (coalesced per channel), fp32 -> bf16 ----
  for (int i = tid; i < MT * C_DIM; i += NTHREADS) {
    int c = i >> 7;            // / 128
    int t = i & (MT - 1);
    tokLds[t * TOK_STRIDE + c] = f2bf(xb[(size_t)c * HW + t]);
  }
  __syncthreads();

  // ---- 2) router logits + top-2 softmax gates (dense E-wide gate table) ----
  if (tid < MT) {
    float lg[E_DIM];
#pragma unroll
    for (int e = 0; e < E_DIM; ++e) lg[e] = router_b[e];
    for (int c = 0; c < C_DIM; ++c) {
      float tv = bf2f(tokLds[tid * TOK_STRIDE + c]);
#pragma unroll
      for (int e = 0; e < E_DIM; ++e) lg[e] += tv * router_w[e * C_DIM + c];
    }
    int i0 = 0;
#pragma unroll
    for (int e = 1; e < E_DIM; ++e) if (lg[e] > lg[i0]) i0 = e;
    int i1 = -1;
#pragma unroll
    for (int e = 0; e < E_DIM; ++e)
      if (e != i0 && (i1 < 0 || lg[e] > lg[i1])) i1 = e;
    float w1 = __expf(lg[i1] - lg[i0]);      // softmax over the two kept logits
    float s  = 1.0f / (1.0f + w1);
#pragma unroll
    for (int e = 0; e < E_DIM; ++e) gatesLds[tid * 4 + e] = 0.0f;
    gatesLds[tid * 4 + i0] = s;
    gatesLds[tid * 4 + i1] = w1 * s;
  }
  __syncthreads();

  // ---- 3) per-wave persistent A fragments for its 16-token M-tile ----
  v16bf af[C_DIM / 32];                      // 6 frags, loaded once
#pragma unroll
  for (int k = 0; k < C_DIM / 32; ++k)
    af[k] = load_afrag_lds(tokLds, wv * 16, TOK_STRIDE, k * 32, lane);

  v8f outAcc[C_DIM / 16];                    // 12 persistent output fragments
#pragma unroll
  for (int n = 0; n < C_DIM / 16; ++n) outAcc[n] = v8f{};

  const int rowA    = wv * 16 + ((lane >> 4) << 3); // accum row base (add r)
  const int colBase = lane & 15;

  // ---- 4) expert loop (barrier-free: each wave owns rows [wv*16, wv*16+16)) ----
  for (int e = 0; e < E_DIM; ++e) {
    const unsigned short* w1p = w1bf + (size_t)e * HID_DIM * C_DIM;
    const float*          b1p = fc1_b + e * HID_DIM;

    // overlap next expert's weight first-touch with this expert's compute
    if (e + 1 < E_DIM) {
      __builtin_prefetch(w1bf + (size_t)(e + 1) * HID_DIM * C_DIM, 0, 1);
      __builtin_prefetch(w2bf + (size_t)(e + 1) * C_DIM * HID_DIM, 0, 1);
    }

    // GEMM1: hid = gelu(tok @ fc1^T + b1), result -> bf16 hid tile in LDS
    for (int n = 0; n < HID_DIM / 16; ++n) {          // 24 N-tiles
      v8f acc = v8f{};
#pragma unroll
      for (int k = 0; k < C_DIM / 32; ++k) {          // 6 K-steps
        v16bf bf = load_bfrag_bf(w1p, n * 16, C_DIM, k * 32, lane);
        acc = WMMA_BF16(af[k], bf, acc);
      }
      int col = n * 16 + colBase;
      float bias = b1p[col];
#pragma unroll
      for (int r = 0; r < 8; ++r) {
        float h = gelu_exact(acc[r] + bias);
        hidLds[(rowA + r) * HID_STRIDE + col] = f2bf(h);
      }
    }

    // Hoist GEMM2 A fragments: 12 K-frags loaded once per expert (same-wave LDS
    // ordering guarantees the GEMM1 stores above are visible).
    v16bf a2[HID_DIM / 32];
#pragma unroll
    for (int k = 0; k < HID_DIM / 32; ++k)
      a2[k] = load_afrag_lds(hidLds, wv * 16, HID_STRIDE, k * 32, lane);

    // GEMM2: y = hid @ fc2^T + b2 ; outAcc += gate_e * y
    const unsigned short* w2p = w2bf + (size_t)e * C_DIM * HID_DIM;
    const float*          b2p = fc2_b + e * C_DIM;
    float gate[8];
#pragma unroll
    for (int r = 0; r < 8; ++r) gate[r] = gatesLds[(rowA + r) * 4 + e];

#pragma unroll
    for (int n = 0; n < C_DIM / 16; ++n) {            // 12 N-tiles
      v8f acc = v8f{};
#pragma unroll
      for (int k = 0; k < HID_DIM / 32; ++k) {        // 12 K-steps
        v16bf bf = load_bfrag_bf(w2p, n * 16, HID_DIM, k * 32, lane);
        acc = WMMA_BF16(a2[k], bf, acc);
      }
      int col = n * 16 + colBase;
      float bias = b2p[col];
#pragma unroll
      for (int r = 0; r < 8; ++r)
        outAcc[n][r] += gate[r] * (acc[r] + bias);
    }
  }

  // ---- 5) transpose out fragments through LDS (reuse hid buffer), then
  //         fully-coalesced residual epilogue: out = x + scale * moe ----
#pragma unroll
  for (int n = 0; n < C_DIM / 16; ++n) {
    int col = n * 16 + colBase;
#pragma unroll
    for (int r = 0; r < 8; ++r)
      outLds[(rowA + r) * OUT_STRIDE + col] = outAcc[n][r];
  }
  __syncthreads();

  const float scale = scale_p[0];
  for (int i = tid; i < MT * C_DIM; i += NTHREADS) {
    int c = i >> 7;
    int t = i & (MT - 1);
    ob[(size_t)c * HW + t] = xb[(size_t)c * HW + t] + scale * outLds[t * OUT_STRIDE + c];
  }
}

extern "C" void kernel_launch(void* const* d_in, const int* in_sizes, int n_in,
                              void* d_out, int out_size, void* d_ws, size_t ws_size,
                              hipStream_t stream) {
  const float* x        = (const float*)d_in[0];
  const float* router_w = (const float*)d_in[1];
  const float* router_b = (const float*)d_in[2];
  const float* fc1_w    = (const float*)d_in[3];
  const float* fc1_b    = (const float*)d_in[4];
  const float* fc2_w    = (const float*)d_in[5];
  const float* fc2_b    = (const float*)d_in[6];
  const float* scale    = (const float*)d_in[7];
  float* out            = (float*)d_out;

  // workspace layout: [w1bf: W_ELEMS bf16][w2bf: W_ELEMS bf16]  (~1.13 MB total)
  unsigned short* w1bf = (unsigned short*)d_ws;
  unsigned short* w2bf = w1bf + W_ELEMS;

  const int cvt_blocks = (W_ELEMS + 255) / 256;
  convert_weights_kernel<<<cvt_blocks, 256, 0, stream>>>(fc1_w, fc2_w, w1bf, w2bf);

  const int blocks = (B_DIM * HW) / MT;                       // 1024 tiles
  const size_t smem_bytes =
      (size_t)MT * (TOK_STRIDE + HID_STRIDE) * sizeof(unsigned short)  // 151552
      + (size_t)MT * 4 * sizeof(float);                                // + 2048 = 153600

  hipFuncSetAttribute(reinterpret_cast<const void*>(moe_topk_kernel),
                      hipFuncAttributeMaxDynamicSharedMemorySize, (int)smem_bytes);

  moe_topk_kernel<<<blocks, NTHREADS, smem_bytes, stream>>>(
      x, router_w, router_b, w1bf, fc1_b, w2bf, fc2_b, scale, out);
}
